// MultiHeadedAttention_90975997264525
// MI455X (gfx1250) — compile-verified, matrix operand order checked by reading, not run on hardware
//
#include <hip/hip_runtime.h>
#include <hip/hip_bf16.h>

// ---------------------------------------------------------------------------
// MultiHeadedAttention for MI455X (gfx1250, wave32, WMMA + TDM).
// B=2, S=2048, D=512, H=8, DK=64.  Bandwidth-bound on the 268MB bias tensor
// (~12us floor at 23.3 TB/s): single streaming pass (flash-attention style),
// f16 WMMA with fp32 accumulation, K/V tiles streamed into LDS by the Tensor
// Data Mover (double buffered, overlapped with WMMA), bias/mask loads marked
// non-temporal.
// ---------------------------------------------------------------------------

typedef __attribute__((ext_vector_type(16))) _Float16 v16h;
typedef __attribute__((ext_vector_type(4)))  _Float16 v4h;
typedef __attribute__((ext_vector_type(8)))  float    v8f;
typedef __attribute__((ext_vector_type(4)))  float    v4f;
typedef __attribute__((ext_vector_type(4)))  unsigned v4u;
typedef __attribute__((ext_vector_type(8)))  int      v8i;
typedef __attribute__((ext_vector_type(4)))  int      v4i;

namespace {
constexpr int B  = 2;
constexpr int S  = 2048;
constexpr int D  = 512;
constexpr int H  = 8;
constexpr int DK = 64;
constexpr float NEG = -60000.0f;
constexpr float QK_SCALE = 0.125f;   // 1/sqrt(64)
constexpr int LD = 80;               // LDS tile row stride (halves): 64 + 16 pad
}

// D = A(16x32 f16) * B(32x16 f16) + C(16x16 f32)
__device__ __forceinline__ v8f wmma_f16(v16h a, v16h b, v8f c) {
  return __builtin_amdgcn_wmma_f32_16x16x32_f16(
      /*neg_a=*/false, a, /*neg_b=*/false, b,
      /*c_mod=*/(short)0, c, /*reuse_a=*/false, /*reuse_b=*/false);
}

// A-fragment (16x32, ISA 7.12.2 layout) from a row-major f16 LDS tile.
// lane<16: row=row0+lane,    halves 0..7 -> K=kk+0..7,  8..15 -> K=kk+16..23
// lane>=16: row=row0+lane-16, halves 0..7 -> K=kk+8..15, 8..15 -> K=kk+24..31
__device__ __forceinline__ v16h frag_ld_rows(const _Float16* base, int stride,
                                             int row0, int kk, int lane) {
  const _Float16* p = base + (row0 + (lane & 15)) * stride + kk + (((lane >> 4) & 1) << 3);
  v16h r;
#pragma unroll
  for (int i = 0; i < 8; ++i) r[i] = p[i];
#pragma unroll
  for (int i = 0; i < 8; ++i) r[i + 8] = p[16 + i];
  return r;
}

// B-fragment (32x16) from a row-major [K][N] f16 LDS tile (strided K reads).
__device__ __forceinline__ v16h frag_ld_cols(const _Float16* base, int stride,
                                             int col0, int kk, int lane) {
  const _Float16* p = base + (kk + (((lane >> 4) & 1) << 3)) * stride + col0 + (lane & 15);
  v16h r;
#pragma unroll
  for (int i = 0; i < 8; ++i) r[i] = p[i * stride];
#pragma unroll
  for (int i = 0; i < 8; ++i) r[i + 8] = p[(16 + i) * stride];
  return r;
}

// ---------------------------------------------------------------------------
// TDM: DMA a 64x64 f16 tile (global row stride = D elements) into LDS with
// row padding 128B data + 32B pad  ->  LDS row stride 160B (LD=80 halves).
// D# built per CDNA5 ISA ch.8 (groups 0/1; groups 2/3 unused for 2D tiles).
// This toolchain's builtin takes 6 args (extra int32x8 group before cpol).
// ---------------------------------------------------------------------------
__device__ __forceinline__ void tdm_load_tile_64x64(const _Float16* gptr,
                                                    unsigned lds_byte_off) {
  const unsigned long long ga = (unsigned long long)(size_t)gptr;
  v4u g0;
  g0[0] = 1u;                                    // count=1, user descriptor
  g0[1] = lds_byte_off;                          // lds_addr (bytes)
  g0[2] = (unsigned)(ga & 0xffffffffu);          // global_addr[31:0]
  g0[3] = (unsigned)((ga >> 32) & 0x01ffffffu)   // global_addr[56:32]
        | (2u << 30);                            // type = 2 ("image")
  v8i g1;
  g1[0] = (1 << 16)                              // data_size: 2 bytes
        | (1 << 20)                              // pad_enable
        | (4 << 22)                              // pad_interval: every 32 DW (128B)
        | (7 << 25);                             // pad_amount: 8 DW (32B)
  g1[1] = (64 << 16);                            // tensor_dim0 = 64 (bits 79:48)
  g1[2] = (64 << 16);                            // tensor_dim1 = 64 (bits 111:80)
  g1[3] = (64 << 16);                            // tile_dim0 = 64 (bits 127:112)
  g1[4] = 64;                                    // tile_dim1 = 64, tile_dim2 = 0
  g1[5] = D;                                     // tensor_dim0_stride = 512
  g1[6] = 0;
  g1[7] = 0;
  const v4i z4 = {};
  const v8i z8 = {};
  __builtin_amdgcn_tensor_load_to_lds(g0, g1, z4, z4, z8, /*cpol=*/0);
}

__device__ __forceinline__ unsigned lds_off(const void* p) {
  // generic-space LDS pointer: low 32 bits are the LDS byte address
  return (unsigned)(size_t)p;
}

// ---------------------------------------------------------------------------
// GEMM: Y[M=4096, 512] = X[4096,512] * W[512,512] + bias.
// Block tile 128(M) x 64(N), 8 waves, each wave 16x64.  K-step = 32.
// ---------------------------------------------------------------------------
template <typename TIN, bool OUT_F32>
__global__ __launch_bounds__(256) void proj_kernel(
    const TIN* __restrict__ X, const float* __restrict__ W,
    const float* __restrict__ bvec, _Float16* __restrict__ Yh,
    float* __restrict__ Yf) {
  constexpr int LDA = 40;  // 32 + pad
  constexpr int LDB = 72;  // 64 + pad
  __shared__ alignas(16) _Float16 As[128 * LDA];
  __shared__ alignas(16) _Float16 Bs[32 * LDB];

  const int t = threadIdx.x;
  const int lane = t & 31;
  const int wave = t >> 5;
  const int mblk = blockIdx.x * 128;
  const int nblk = blockIdx.y * 64;

  v8f acc[4] = {};

  for (int kk = 0; kk < D; kk += 32) {
    __syncthreads();
    // A tile 128x32 (1024 4-elem chunks, 4 per thread), convert to f16
#pragma unroll
    for (int i = 0; i < 4; ++i) {
      int idx = t + i * 256;
      int r = idx >> 3, c4 = (idx & 7) * 4;
      v4h hx;
      if constexpr (__is_same(TIN, float)) {
        v4f x = *(const v4f*)(X + (size_t)(mblk + r) * D + kk + c4);
#pragma unroll
        for (int j = 0; j < 4; ++j) hx[j] = (_Float16)x[j];
      } else {
        hx = *(const v4h*)(X + (size_t)(mblk + r) * D + kk + c4);
      }
      *(v4h*)(As + r * LDA + c4) = hx;
    }
    // B tile 32x64 (512 4-elem chunks, 2 per thread); W is [in,out]
#pragma unroll
    for (int i = 0; i < 2; ++i) {
      int idx = t + i * 256;
      int r = idx >> 4, c4 = (idx & 15) * 4;
      v4f w = *(const v4f*)(W + (size_t)(kk + r) * D + nblk + c4);
      v4h hw;
#pragma unroll
      for (int j = 0; j < 4; ++j) hw[j] = (_Float16)w[j];
      *(v4h*)(Bs + r * LDB + c4) = hw;
    }
    __syncthreads();

    v16h a = frag_ld_rows(As, LDA, wave * 16, 0, lane);
#pragma unroll
    for (int nt = 0; nt < 4; ++nt) {
      v16h b = frag_ld_cols(Bs, LDB, nt * 16, 0, lane);
      acc[nt] = wmma_f16(a, b, acc[nt]);
    }
  }

  // C layout: VGPR r -> row (base + r) lanes 0..15, row (base+8+r) lanes 16..31
  const int m0 = mblk + wave * 16 + (((lane >> 4) & 1) << 3);
  const int nc = nblk + (lane & 15);
#pragma unroll
  for (int nt = 0; nt < 4; ++nt) {
    const int n = nc + nt * 16;
    const float bb = bvec[n];
#pragma unroll
    for (int r = 0; r < 8; ++r) {
      float val = acc[nt][r] + bb;
      if constexpr (OUT_F32) Yf[(size_t)(m0 + r) * D + n] = val;
      else                   Yh[(size_t)(m0 + r) * D + n] = (_Float16)val;
    }
  }
}

// ---------------------------------------------------------------------------
// Flash attention: one block per (b, h, 64-row q tile). 4 waves; each wave
// owns 16 q rows. 64-key tiles streamed into double-buffered LDS by the TDM
// (issued by wave 0, overlapped with WMMA); bias/mask fused (non-temporal);
// online softmax with xor-shuffle row reductions.
// ---------------------------------------------------------------------------
__global__ __launch_bounds__(128) void attn_kernel(
    const _Float16* __restrict__ Qh, const _Float16* __restrict__ Kh,
    const _Float16* __restrict__ Vh, const int* __restrict__ mask,
    const float* __restrict__ bias, _Float16* __restrict__ AO) {
  __shared__ alignas(16) _Float16 sQ[64 * LD];
  __shared__ alignas(16) _Float16 sP[64 * LD];
  __shared__ alignas(16) _Float16 sK[2][64 * LD];
  __shared__ alignas(16) _Float16 sV[2][64 * LD];

  const int t = threadIdx.x;
  const int lane = t & 31;
  const int wave = t >> 5;
  const int qt = blockIdx.x, h = blockIdx.y, b = blockIdx.z;
  const int qbase = qt * 64;
  constexpr int NT = S / 64;

  const size_t kv_head = (size_t)b * S * D + h * DK;  // + row*D + col

  // Kick off TDM for the first K/V tiles (wave 0 owns all TDM traffic).
  if (wave == 0) {
    tdm_load_tile_64x64(Kh + kv_head, lds_off(sK[0]));
    tdm_load_tile_64x64(Vh + kv_head, lds_off(sV[0]));
  }

  // Stage the 64x64 Q tile (f16, 16B vector loads).
#pragma unroll
  for (int i = 0; i < 4; ++i) {
    int chunk = t + i * 128;                 // 512 chunks of 8 halves
    int r = chunk >> 3, c8 = (chunk & 7) * 8;
    *(uint4*)(sQ + r * LD + c8) =
        *(const uint4*)(Qh + (size_t)(b * S + qbase + r) * D + h * DK + c8);
  }
  __syncthreads();

  const int m0 = wave * 16;
  const v16h aq0 = frag_ld_rows(sQ, LD, m0, 0, lane);
  const v16h aq1 = frag_ld_rows(sQ, LD, m0, 32, lane);

  v8f oacc[4] = {};
  float mrow[8], lrow[8];
#pragma unroll
  for (int r = 0; r < 8; ++r) { mrow[r] = -3.0e38f; lrow[r] = 0.0f; }

  const int qloc = m0 + (((lane >> 4) & 1) << 3);  // +r = local q row of acc elem
  const int ncol = lane & 15;

  for (int kt = 0; kt < NT; ++kt) {
    const int kbase = kt * 64;
    const int cur = kt & 1;

    if (wave == 0) __builtin_amdgcn_s_wait_tensorcnt(0);  // tile kt landed
    __syncthreads();  // all waves done with buffer cur^1 (read in iter kt-1)
    if (wave == 0 && kt + 1 < NT) {   // overlap next tile's DMA with compute
      const size_t nxt = kv_head + (size_t)(kbase + 64) * D;
      tdm_load_tile_64x64(Kh + nxt, lds_off(sK[cur ^ 1]));
      tdm_load_tile_64x64(Vh + nxt, lds_off(sV[cur ^ 1]));
    }

    // scores = Q * K^T  (K tile is the transposed operand -> A-style reads)
    v8f sc[4] = {};
#pragma unroll
    for (int nt = 0; nt < 4; ++nt) {
      v16h bk0 = frag_ld_rows(sK[cur], LD, nt * 16, 0, lane);
      sc[nt] = wmma_f16(aq0, bk0, sc[nt]);
      v16h bk1 = frag_ld_rows(sK[cur], LD, nt * 16, 32, lane);
      sc[nt] = wmma_f16(aq1, bk1, sc[nt]);
    }

    // fused scale + bias + mask (bias/mask are single-use: non-temporal)
    float sv[4][8];
#pragma unroll
    for (int nt = 0; nt < 4; ++nt) {
#pragma unroll
      for (int r = 0; r < 8; ++r) {
        const int qg = qbase + qloc + r;
        const int kg = kbase + nt * 16 + ncol;
        const float bval =
            __builtin_nontemporal_load(bias + (((size_t)b * H + h) * S + qg) * S + kg);
        const int mval =
            __builtin_nontemporal_load(mask + ((size_t)b * S + qg) * S + kg);
        float val = sc[nt][r] * QK_SCALE + bval;
        if (mval == 0) val = NEG;
        sv[nt][r] = val;
      }
    }

    // online softmax; rows live in 16-lane halves -> xor-shuffle reductions
#pragma unroll
    for (int r = 0; r < 8; ++r) {
      float rmax = fmaxf(fmaxf(sv[0][r], sv[1][r]), fmaxf(sv[2][r], sv[3][r]));
#pragma unroll
      for (int off = 1; off < 16; off <<= 1)
        rmax = fmaxf(rmax, __shfl_xor(rmax, off, 32));
      const float mnew = fmaxf(mrow[r], rmax);
      const float alpha = __expf(mrow[r] - mnew);
      float rsum = 0.0f;
#pragma unroll
      for (int nt = 0; nt < 4; ++nt) {
        const float p = __expf(sv[nt][r] - mnew);
        sv[nt][r] = p;
        rsum += p;
      }
#pragma unroll
      for (int off = 1; off < 16; off <<= 1)
        rsum += __shfl_xor(rsum, off, 32);
      lrow[r] = lrow[r] * alpha + rsum;
      mrow[r] = mnew;
#pragma unroll
      for (int nt = 0; nt < 4; ++nt) oacc[nt][r] *= alpha;
      // P -> LDS (f16) for re-read in A-fragment layout
#pragma unroll
      for (int nt = 0; nt < 4; ++nt)
        sP[(qloc + r) * LD + nt * 16 + ncol] = (_Float16)sv[nt][r];
    }
    __syncthreads();

    // O += P * V
#pragma unroll
    for (int kk = 0; kk < 64; kk += 32) {
      v16h ap = frag_ld_rows(sP, LD, m0, kk, lane);
#pragma unroll
      for (int nt = 0; nt < 4; ++nt) {
        v16h bv = frag_ld_cols(sV[cur], LD, nt * 16, kk, lane);
        oacc[nt] = wmma_f16(ap, bv, oacc[nt]);
      }
    }
  }

  // normalize and write merged-head layout [B,S,D] (f16 -> O-projection input)
#pragma unroll
  for (int nt = 0; nt < 4; ++nt) {
#pragma unroll
    for (int r = 0; r < 8; ++r) {
      const int qg = qbase + qloc + r;
      AO[(size_t)(b * S + qg) * D + h * DK + nt * 16 + ncol] =
          (_Float16)(oacc[nt][r] / lrow[r]);
    }
  }
}

// ---------------------------------------------------------------------------
extern "C" void kernel_launch(void* const* d_in, const int* in_sizes, int n_in,
                              void* d_out, int out_size, void* d_ws, size_t ws_size,
                              hipStream_t stream) {
  (void)in_sizes; (void)n_in; (void)out_size; (void)ws_size;
  const float* q    = (const float*)d_in[0];
  const float* k    = (const float*)d_in[1];
  const float* v    = (const float*)d_in[2];
  const int*   mask = (const int*)d_in[3];
  const float* bias = (const float*)d_in[4];
  const float* Wq = (const float*)d_in[5];  const float* bq = (const float*)d_in[6];
  const float* Wk = (const float*)d_in[7];  const float* bk = (const float*)d_in[8];
  const float* Wv = (const float*)d_in[9];  const float* bv = (const float*)d_in[10];
  const float* Wo = (const float*)d_in[11]; const float* bo = (const float*)d_in[12];
  float* out = (float*)d_out;

  const size_t MAT = (size_t)B * S * D;  // 2M elements, f16 -> 4MB each
  _Float16* Qh  = (_Float16*)d_ws;
  _Float16* Kh  = Qh + MAT;
  _Float16* Vh  = Kh + MAT;
  _Float16* AOh = Vh + MAT;

  dim3 gp(B * S / 128, D / 64, 1);   // 32 x 8 blocks, 256 threads
  proj_kernel<float, false><<<gp, 256, 0, stream>>>(q, Wq, bq, Qh, nullptr);
  proj_kernel<float, false><<<gp, 256, 0, stream>>>(k, Wk, bk, Kh, nullptr);
  proj_kernel<float, false><<<gp, 256, 0, stream>>>(v, Wv, bv, Vh, nullptr);

  dim3 ga(S / 64, H, B);             // 32 x 8 x 2 blocks, 128 threads
  attn_kernel<<<ga, 128, 0, stream>>>(Qh, Kh, Vh, mask, bias, AOh);

  proj_kernel<_Float16, true><<<gp, 256, 0, stream>>>(AOh, Wo, bo, nullptr, out);
}
